// CausalSelfAttention_40046275068084
// MI455X (gfx1250) — matrix operand order repaired and checked by reading, not compiled, not run on hardware
//
#include <hip/hip_runtime.h>

// ---------------------------------------------------------------------------
// CausalSelfAttention for MI455X (gfx1250, wave32, WMMA + TDM).
// B=4, T=2048, C=1024, H=16, hd=64 (fixed by the reference).
//
//   0) cvt_f32_bf16: one-time x / w_attn / w_proj -> bf16 (memory-bound).
//   1) qkv_gemm  : bf16 TDM-staged 128x128x64 tiles, double-buffered;
//                  epilogue scatters Q,K [B,H,T,64] and V^T [B,H,64,T] + bias.
//   2) flash_attn: block-uniform 64-key blocks, K/V staged by double-buffered
//                  TDM into LDS; online softmax; WMMA for QK^T and P*V.
//   3) proj_gemm : y(bf16) x w_proj^T + b_proj -> out (f32), same pipeline.
//
// All wave-level gates (TDM issue, past-diagonal skip) go through
// readfirstlane so they compile to scalar branches: TDM ignores EXEC and
// WMMA is never EXEC-skipped, so these regions must be branched over, not
// lane-predicated.
//
// Workspace (bf16): Q | K | V^T | {x-bf16 then Y} | w_attn-bf16 | w_proj-bf16
//                 = 16+16+16+16+6+2 = 72 MB.
// ---------------------------------------------------------------------------

typedef __bf16 bf16;
typedef __attribute__((ext_vector_type(16))) __bf16 v16bf;
typedef __attribute__((ext_vector_type(8)))  float  v8f;
typedef __attribute__((ext_vector_type(4)))  int    v4i;
typedef __attribute__((ext_vector_type(8)))  int    v8i;

__device__ __forceinline__ v8f wmma_bf16(v16bf a, v16bf b, v8f c) {
  return __builtin_amdgcn_wmma_f32_16x16x32_bf16(
      false, a, false, b, (short)0, c, false, false);
}

// Wave-uniform scalar "am I wave 0 of the block" -> forces s_cmp/s_cbranch.
__device__ __forceinline__ bool is_wave0() {
  return __builtin_amdgcn_readfirstlane((int)threadIdx.x) < 32;
}

// TDM: DMA a 2-D tile (tile_d0 x tile_d1, contiguous dim0) global -> LDS.
// One scalar-operand instruction, tracked by TENSORcnt (in-order per wave).
// D# per CDNA5 ISA; data_size code 1=2B.
__device__ __forceinline__ void tdm_load_2d(unsigned lds_addr, const void* gaddr,
                                            unsigned ds_code, unsigned tile_d0,
                                            unsigned tile_d1, unsigned tensor_d0,
                                            unsigned tensor_d1, unsigned stride0) {
  const unsigned long long ga = (unsigned long long)gaddr;
  v4i g0;
  g0[0] = 1;                                      // count=1, user mode, no gather
  g0[1] = (int)lds_addr;
  g0[2] = (int)(unsigned)(ga & 0xffffffffu);
  g0[3] = (int)(((unsigned)(ga >> 32) & 0x01ffffffu) | (2u << 30));  // type=2
  v8i g1;
  g1[0] = (int)(ds_code << 16);
  g1[1] = (int)(tensor_d0 << 16);
  g1[2] = (int)((tensor_d0 >> 16) | (tensor_d1 << 16));
  g1[3] = (int)((tensor_d1 >> 16) | (tile_d0 << 16));
  g1[4] = (int)tile_d1;
  g1[5] = (int)stride0;
  g1[6] = 0;
  g1[7] = 0;
  asm volatile("tensor_load_to_lds %0, %1" :: "s"(g0), "s"(g1) : "memory");
}

__device__ __forceinline__ unsigned lds_addr_of(const void* p) {
  return (unsigned)(unsigned long long)p;  // flat shared addr: LDS off in [31:0]
}

// A-operand 16x32 bf16 (row-major src): lane l row m=l&15, half g=l>>4;
// VGPR r<4 -> K=2r(+8g), r>=4 -> K=16+2(r-4)(+8g).   (ISA 7.12.2)
__device__ __forceinline__ v16bf load_A(const bf16* src, int ld) {
  const int lane = threadIdx.x & 31;
  const bf16* p = src + (lane & 15) * ld + 8 * (lane >> 4);
  v16bf a;
#pragma unroll
  for (int r = 0; r < 8; ++r) {
    const int kp = ((r & 4) << 2) + 2 * (r & 3);
    a[2 * r] = p[kp]; a[2 * r + 1] = p[kp + 1];
  }
  return a;
}
// B-operand 32x16 bf16 (column-major src: (k,n) at n*ld+k): lane l col n=l&15,
// element i<8 -> K=8*(l>>4)+i, i>=8 -> K=16+8*(l>>4)+(i-8).
__device__ __forceinline__ v16bf load_B(const bf16* src, int ld) {
  const int lane = threadIdx.x & 31;
  const bf16* p = src + (lane & 15) * ld + 8 * (lane >> 4);
  v16bf b;
#pragma unroll
  for (int i = 0; i < 16; ++i) b[i] = p[i < 8 ? i : i + 8];
  return b;
}

// ---------------------------------------------------------------------------
// Kernel 0: f32 -> bf16 bulk convert (float4 in, 4x bf16 out).
// ---------------------------------------------------------------------------
__global__ __launch_bounds__(256) void cvt_f32_bf16(
    const float4* __restrict__ src, uint2* __restrict__ dst, int n4) {
  const int i = blockIdx.x * blockDim.x + threadIdx.x;
  if (i >= n4) return;
  const float4 f = src[i];
  union { bf16 h[4]; uint2 u; } r;
  r.h[0] = (bf16)f.x; r.h[1] = (bf16)f.y; r.h[2] = (bf16)f.z; r.h[3] = (bf16)f.w;
  dst[i] = r.u;
}

// ---------------------------------------------------------------------------
// Kernel 1: QKV GEMM.  grid(64,24), block 256 (8 waves, 2x4 wave grid).
// Block tile 128x128, K-step 64, double-buffered TDM staging.
// ---------------------------------------------------------------------------
__global__ __launch_bounds__(256) void qkv_gemm(
    const bf16* __restrict__ xb, const bf16* __restrict__ wab,
    const float* __restrict__ bA, bf16* __restrict__ qw,
    bf16* __restrict__ kw, bf16* __restrict__ vtw) {
  __shared__ bf16 sA[2][128 * 64];  // [m][k]
  __shared__ bf16 sB[2][128 * 64];  // [n][k]
  const int tid = threadIdx.x;
  const int m0 = blockIdx.x * 128, n0 = blockIdx.y * 128;
  const int w = tid >> 5, lane = tid & 31;
  const int nl = lane & 15, g = lane >> 4;
  const int wm = (w >> 2) * 64, wn = (w & 3) * 32;
  const bool wave0 = is_wave0();

  auto issue = [&](int it, int buf) {
    tdm_load_2d(lds_addr_of(sA[buf]), xb + (size_t)m0 * 1024 + it * 64,
                1, 64, 128, 1024, 8192, 1024);
    tdm_load_2d(lds_addr_of(sB[buf]), wab + (size_t)n0 * 1024 + it * 64,
                1, 64, 128, 1024, 3072, 1024);
  };

  v8f acc[4][2];
#pragma unroll
  for (int mt = 0; mt < 4; ++mt)
#pragma unroll
    for (int nt = 0; nt < 2; ++nt) { v8f z = {}; acc[mt][nt] = z; }

  if (wave0) issue(0, 0);
  for (int it = 0; it < 16; ++it) {
    if (wave0) {
      if (it + 1 < 16) { issue(it + 1, (it + 1) & 1);
                         __builtin_amdgcn_s_wait_tensorcnt(2); }
      else             { __builtin_amdgcn_s_wait_tensorcnt(0); }
    }
    __syncthreads();
    const bf16* a = sA[it & 1];
    const bf16* bm = sB[it & 1];
#pragma unroll
    for (int kc = 0; kc < 2; ++kc) {
      v16bf af[4], bfb[2];
#pragma unroll
      for (int mt = 0; mt < 4; ++mt)
        af[mt] = load_A(a + (wm + mt * 16) * 64 + kc * 32, 64);
#pragma unroll
      for (int nt = 0; nt < 2; ++nt)
        bfb[nt] = load_B(bm + (wn + nt * 16) * 64 + kc * 32, 64);
#pragma unroll
      for (int mt = 0; mt < 4; ++mt)
#pragma unroll
        for (int nt = 0; nt < 2; ++nt)
          acc[mt][nt] = wmma_bf16(af[mt], bfb[nt], acc[mt][nt]);
    }
    __syncthreads();
  }

  // Epilogue: bias, bf16, scatter Q [B,H,T,64], K [B,H,T,64], V^T [B,H,64,T].
#pragma unroll
  for (int nt = 0; nt < 2; ++nt) {
    const int f = n0 + wn + nt * 16 + nl;
    const float bias = bA[f];
    const int sec = f >> 10, c = f & 1023, head = c >> 6, d = c & 63;
#pragma unroll
    for (int mt = 0; mt < 4; ++mt)
#pragma unroll
      for (int r = 0; r < 8; ++r) {
        const int row = m0 + wm + mt * 16 + r + 8 * g;  // global b*T+t
        const int bb = row >> 11, t = row & 2047;
        const bf16 v = (bf16)(acc[mt][nt][r] + bias);
        const size_t bh = (size_t)bb * 16 + head;
        if (sec == 0)      qw[(bh * 2048 + t) * 64 + d] = v;
        else if (sec == 1) kw[(bh * 2048 + t) * 64 + d] = v;
        else               vtw[(bh * 64 + d) * 2048 + t] = v;
      }
  }
}

// ---------------------------------------------------------------------------
// Kernel 2: flash attention.  grid(16, 64), block 256.
// Block-uniform 64-key blocks, K/V staged by double-buffered TDM.
// Wave w owns q rows [q0+16w, +16); waves branch off past their diagonal.
// ---------------------------------------------------------------------------
__global__ __launch_bounds__(256) void flash_attn(
    const bf16* __restrict__ q, const bf16* __restrict__ k,
    const bf16* __restrict__ vt, bf16* __restrict__ y) {
  constexpr int T = 2048, HD = 64;
  __shared__ bf16 sK[2][64 * 64];   // [key][hd]
  __shared__ bf16 sV[2][64 * 64];   // [hd][key]
  __shared__ bf16 sP[8 * 16 * 64];  // wave-private P tiles

  const int tid = threadIdx.x;
  const int bh = blockIdx.y, b = bh >> 4, h = bh & 15;
  const int q0 = blockIdx.x * 128;
  const int w = tid >> 5, lane = tid & 31;
  const int nl = lane & 15, g = lane >> 4;
  const int qrow = q0 + w * 16;
  const bool wave0 = is_wave0();

  const bf16* qb = q  + ((size_t)bh * T + qrow) * HD;
  const bf16* kb = k  + (size_t)bh * T * HD;
  const bf16* vb = vt + (size_t)bh * HD * T;

  auto issue = [&](int j, int buf) {
    tdm_load_2d(lds_addr_of(sK[buf]), kb + (size_t)j * 64 * HD,
                1, 64, 64, 64, 2048, 64);          // contiguous 64x64 K block
    tdm_load_2d(lds_addr_of(sV[buf]), vb + (size_t)j * 64,
                1, 64, 64, 2048, 64, 2048);        // strided 64x64 V^T tile
  };

  const v16bf aq0 = load_A(qb, HD);
  const v16bf aq1 = load_A(qb + 32, HD);

  v8f o[4];
  float rm[8], rs[8];
#pragma unroll
  for (int i = 0; i < 4; ++i) { v8f z = {}; o[i] = z; }
#pragma unroll
  for (int r = 0; r < 8; ++r) { rm[r] = -1e30f; rs[r] = 0.f; }

  const int jblocks = (q0 >> 6) + 2;  // key blocks this 128-row q chunk needs
  // Wave-uniform scalar last-block index -> true s_cbranch around compute
  // (WMMA is never EXEC-skipped, so the skip must be a real branch).
  const int wjmax = __builtin_amdgcn_readfirstlane(qrow) >> 6;

  if (wave0) issue(0, 0);
  for (int j = 0; j < jblocks; ++j) {
    if (wave0) {
      if (j + 1 < jblocks) { issue(j + 1, (j + 1) & 1);
                             __builtin_amdgcn_s_wait_tensorcnt(2); }
      else                 { __builtin_amdgcn_s_wait_tensorcnt(0); }
    }
    __syncthreads();
    if (j <= wjmax) {
      const int kbase = j * 64;
      const bf16* kt = sK[j & 1];
      const bf16* vte = sV[j & 1];

      v8f s[4];
#pragma unroll
      for (int nt = 0; nt < 4; ++nt) {
        const v16bf b0 = load_B(kt + (nt * 16) * 64, 64);
        const v16bf b1 = load_B(kt + (nt * 16) * 64 + 32, 64);
        v8f a = {};
        a = wmma_bf16(aq0, b0, a);
        a = wmma_bf16(aq1, b1, a);
        s[nt] = a;
      }
#pragma unroll
      for (int nt = 0; nt < 4; ++nt) {
        const int kg = kbase + nt * 16 + nl;
#pragma unroll
        for (int r = 0; r < 8; ++r) {
          float v = s[nt][r] * 0.125f;           // 1/sqrt(64)
          if (kg > qrow + r + 8 * g) v = -1e30f; // causal mask
          s[nt][r] = v;
        }
      }

      // Online softmax; row stats live across each 16-lane half.
#pragma unroll
      for (int r = 0; r < 8; ++r) {
        float pm = fmaxf(fmaxf(s[0][r], s[1][r]), fmaxf(s[2][r], s[3][r]));
#pragma unroll
        for (int mk = 1; mk < 16; mk <<= 1) pm = fmaxf(pm, __shfl_xor(pm, mk, 16));
        const float nm = fmaxf(rm[r], pm);
        const float corr = __expf(rm[r] - nm);
        rm[r] = nm;
        float ps = 0.f;
#pragma unroll
        for (int nt = 0; nt < 4; ++nt) {
          const float p = __expf(s[nt][r] - nm);
          s[nt][r] = p; ps += p;
        }
#pragma unroll
        for (int mk = 1; mk < 16; mk <<= 1) ps += __shfl_xor(ps, mk, 16);
        rs[r] = rs[r] * corr + ps;
#pragma unroll
        for (int ot = 0; ot < 4; ++ot) o[ot][r] *= corr;
      }

      // P (C-layout) -> wave-private LDS -> A-layout for P*V.
      bf16* p = sP + w * 16 * 64;
#pragma unroll
      for (int nt = 0; nt < 4; ++nt)
#pragma unroll
        for (int r = 0; r < 8; ++r)
          p[(r + 8 * g) * 64 + nt * 16 + nl] = (bf16)s[nt][r];
      asm volatile("s_wait_dscnt 0" ::: "memory");
      const v16bf pa0 = load_A(p, 64);
      const v16bf pa1 = load_A(p + 32, 64);

#pragma unroll
      for (int ot = 0; ot < 4; ++ot) {
        const v16bf bv0 = load_B(vte + (ot * 16) * 64, 64);
        const v16bf bv1 = load_B(vte + (ot * 16) * 64 + 32, 64);
        o[ot] = wmma_bf16(pa0, bv0, o[ot]);
        o[ot] = wmma_bf16(pa1, bv1, o[ot]);
      }
    }
    __syncthreads();
  }

#pragma unroll
  for (int r = 0; r < 8; ++r) rs[r] = 1.f / rs[r];
  bf16* yb = y + ((size_t)b * T + qrow) * 1024 + h * 64;
#pragma unroll
  for (int ot = 0; ot < 4; ++ot)
#pragma unroll
    for (int r = 0; r < 8; ++r)
      yb[(size_t)(r + 8 * g) * 1024 + ot * 16 + nl] = (bf16)(o[ot][r] * rs[r]);
}

// ---------------------------------------------------------------------------
// Kernel 3: output projection GEMM.  grid(64, 8), block 256.  Out = f32.
// ---------------------------------------------------------------------------
__global__ __launch_bounds__(256) void proj_gemm(
    const bf16* __restrict__ yw, const bf16* __restrict__ wpb,
    const float* __restrict__ bP, float* __restrict__ out) {
  __shared__ bf16 sA[2][128 * 64];
  __shared__ bf16 sB[2][128 * 64];
  const int tid = threadIdx.x;
  const int m0 = blockIdx.x * 128, n0 = blockIdx.y * 128;
  const int w = tid >> 5, lane = tid & 31;
  const int nl = lane & 15, g = lane >> 4;
  const int wm = (w >> 2) * 64, wn = (w & 3) * 32;
  const bool wave0 = is_wave0();

  auto issue = [&](int it, int buf) {
    tdm_load_2d(lds_addr_of(sA[buf]), yw + (size_t)m0 * 1024 + it * 64,
                1, 64, 128, 1024, 8192, 1024);
    tdm_load_2d(lds_addr_of(sB[buf]), wpb + (size_t)n0 * 1024 + it * 64,
                1, 64, 128, 1024, 1024, 1024);
  };

  v8f acc[4][2];
#pragma unroll
  for (int mt = 0; mt < 4; ++mt)
#pragma unroll
    for (int nt = 0; nt < 2; ++nt) { v8f z = {}; acc[mt][nt] = z; }

  if (wave0) issue(0, 0);
  for (int it = 0; it < 16; ++it) {
    if (wave0) {
      if (it + 1 < 16) { issue(it + 1, (it + 1) & 1);
                         __builtin_amdgcn_s_wait_tensorcnt(2); }
      else             { __builtin_amdgcn_s_wait_tensorcnt(0); }
    }
    __syncthreads();
    const bf16* a = sA[it & 1];
    const bf16* bm = sB[it & 1];
#pragma unroll
    for (int kc = 0; kc < 2; ++kc) {
      v16bf af[4], bfb[2];
#pragma unroll
      for (int mt = 0; mt < 4; ++mt)
        af[mt] = load_A(a + (wm + mt * 16) * 64 + kc * 32, 64);
#pragma unroll
      for (int nt = 0; nt < 2; ++nt)
        bfb[nt] = load_B(bm + (wn + nt * 16) * 64 + kc * 32, 64);
#pragma unroll
      for (int mt = 0; mt < 4; ++mt)
#pragma unroll
        for (int nt = 0; nt < 2; ++nt)
          acc[mt][nt] = wmma_bf16(af[mt], bfb[nt], acc[mt][nt]);
    }
    __syncthreads();
  }

#pragma unroll
  for (int nt = 0; nt < 2; ++nt) {
    const int f = n0 + wn + nt * 16 + nl;
    const float bias = bP[f];
#pragma unroll
    for (int mt = 0; mt < 4; ++mt)
#pragma unroll
      for (int r = 0; r < 8; ++r) {
        const int row = m0 + wm + mt * 16 + r + 8 * g;
        out[(size_t)row * 1024 + f] = acc[mt][nt][r] + bias;
      }
  }
}

// ---------------------------------------------------------------------------
extern "C" void kernel_launch(void* const* d_in, const int* in_sizes, int n_in,
                              void* d_out, int out_size, void* d_ws, size_t ws_size,
                              hipStream_t stream) {
  (void)in_sizes; (void)n_in; (void)out_size; (void)ws_size;
  const float* x      = (const float*)d_in[0];
  const float* w_attn = (const float*)d_in[1];
  const float* b_attn = (const float*)d_in[2];
  const float* w_proj = (const float*)d_in[3];
  const float* b_proj = (const float*)d_in[4];
  float* out = (float*)d_out;

  constexpr size_t SZ  = (size_t)4 * 2048 * 1024;  // B*T*C
  constexpr size_t WAB = (size_t)3 * 1024 * 1024;  // 3C*C
  constexpr size_t WPB = (size_t)1024 * 1024;      // C*C
  bf16* qw   = (bf16*)d_ws;
  bf16* kw   = qw + SZ;
  bf16* vtw  = kw + SZ;
  bf16* xb_y = vtw + SZ;        // x-bf16 during qkv_gemm, then Y
  bf16* wab  = xb_y + SZ;
  bf16* wpb  = wab + WAB;       // 72 MB total

  cvt_f32_bf16<<<(int)(SZ / 4 + 255) / 256, 256, 0, stream>>>(
      (const float4*)x, (uint2*)xb_y, (int)(SZ / 4));
  cvt_f32_bf16<<<(int)(WAB / 4 + 255) / 256, 256, 0, stream>>>(
      (const float4*)w_attn, (uint2*)wab, (int)(WAB / 4));
  cvt_f32_bf16<<<(int)(WPB / 4 + 255) / 256, 256, 0, stream>>>(
      (const float4*)w_proj, (uint2*)wpb, (int)(WPB / 4));

  qkv_gemm<<<dim3(64, 24), 256, 0, stream>>>(xb_y, wab, b_attn, qw, kw, vtw);
  flash_attn<<<dim3(16, 64), 256, 0, stream>>>(qw, kw, vtw, xb_y);
  proj_gemm<<<dim3(64, 8), 256, 0, stream>>>(xb_y, wpb, b_proj, out);
}
